// DotProductionAttention_12189117186184
// MI455X (gfx1250) — matrix operand order repaired and checked by reading, not compiled, not run on hardware
//
#include <hip/hip_runtime.h>

typedef _Float16 half16 __attribute__((ext_vector_type(16)));
typedef _Float16 halfv8 __attribute__((ext_vector_type(8)));
typedef _Float16 half2t __attribute__((ext_vector_type(2)));
typedef __fp16   fp16x2 __attribute__((ext_vector_type(2)));
typedef float    f32x8  __attribute__((ext_vector_type(8)));

#define TQ 64
#define TK 64
#define DHEAD 64
#define KPITCH 72   // K tile: 144B rows, 16B-aligned -> ds_load_b128 fragments
#define VPITCH 66   // VT tile: 132B rows, conflict-free transposed b16 stores

struct MaskOff { static constexpr bool value = false; };
struct MaskOn  { static constexpr bool value = true;  };

// pack two f32 -> half2 (v_cvt_pk_rtz_f16_f32), re-typed to _Float16x2
__device__ __forceinline__ half2t pack_f16(float x, float y) {
  fp16x2 v = __builtin_amdgcn_cvt_pkrtz(x, y);
  return __builtin_bit_cast(half2t, v);
}

// 16x32 f16 A-fragment, base must point at (row + 8h), 16B-aligned.
// elements 0..7 = halves [base..base+7] (K = 8h+e), 8..15 = [base+16..23] (K = 16+8h+e)
__device__ __forceinline__ half16 load_afrag128(const _Float16* base) {
  halfv8 lo = *(const halfv8*)(base);
  halfv8 hi = *(const halfv8*)(base + 16);
  half16 a;
#pragma unroll
  for (int i = 0; i < 8; ++i) { a[i] = lo[i]; a[i + 8] = hi[i]; }
  return a;
}

// 32x16 f16 B-fragment (base at 8h): e<8 <-> K = 8h + e, e>=8 <-> K = 16 + 8h + (e-8)
__device__ __forceinline__ half16 load_bfrag(const _Float16* p) {
  half16 v;
#pragma unroll
  for (int j = 0; j < 8; ++j) {
    half2t t = *(const half2t*)(p + 2 * j + (j >> 2) * 8);
    v[2 * j] = t[0];
    v[2 * j + 1] = t[1];
  }
  return v;
}

__global__ __launch_bounds__(128) void flash_attn_wmma_f16(
    const float* __restrict__ Q, const float* __restrict__ K,
    const float* __restrict__ V, const int* __restrict__ valid_lens,
    float* __restrict__ O, int Nq, int Nk) {
  __shared__ _Float16 Kl[TK * KPITCH];      // K tile, row-major [key][d], f16
  __shared__ _Float16 VTl[DHEAD * VPITCH];  // V tile transposed [d][key], f16

  const int tid  = threadIdx.x;
  const int wave = tid >> 5;
  const int lane = tid & 31;
  const int n    = lane & 15;  // q row for A frags / d col for B frags
  const int h    = lane >> 4;  // half-wave id

  const int b    = blockIdx.y;
  const int q0   = blockIdx.x * TQ;
  const int qrow = q0 + wave * 16 + n;

  const int valid = valid_lens[b];
  const size_t bQ = (size_t)b * Nq * DHEAD;
  const size_t bK = (size_t)b * Nk * DHEAD;

  // cooperative-loader decomposition: only 2 global pointers, constant offsets
  const int c_row = tid >> 5;        // 0..3
  const int dp    = (tid & 31) * 2;  // even 0..62

  // single per-lane fragment bases; all per-(t,chunk) offsets are immediates
  const _Float16* kAbase = &Kl[n * KPITCH + 8 * h];              // + t*16*KPITCH + c*32
  const _Float16* vBbase = &VTl[n * VPITCH + 8 * h];             // + dt*16*VPITCH + kc*32

  // Q fragments: B operand of S^T = K * Q^T. Scale 1/sqrt(64)=0.125 folded (exact).
  half16 qf[2];
  {
    const float* qptr = Q + bQ + (size_t)qrow * DHEAD;
#pragma unroll
    for (int c = 0; c < 2; ++c)
#pragma unroll
      for (int e = 0; e < 16; ++e)
        qf[c][e] = (_Float16)(qptr[c * 32 + (e >> 3) * 16 + h * 8 + (e & 7)] * 0.125f);
  }

  f32x8 acc[4];  // acc[dt][r] = O[q = r + 8h][d = dt*16 + n] (unnormalized)
#pragma unroll
  for (int dt = 0; dt < 4; ++dt) acc[dt] = (f32x8){0.f,0.f,0.f,0.f,0.f,0.f,0.f,0.f};
  float mrow = -INFINITY;  // running max for q = n (identical in both halves)
  float lrow = 0.f;        // running sum

  const int cap = Nk / TK;
  int nfull = valid >> 6;
  if (nfull > cap) nfull = cap;
  const bool has_tail = ((valid & (TK - 1)) != 0) && (nfull < cap);

  auto tile_body = [&](int k0, auto MASKED_TAG) {
    constexpr bool MASKED = decltype(MASKED_TAG)::value;
    // ---- cooperative tile load: K row-major f16, V transposed f16 ----
    {
      const float* kgp = K + bK + ((size_t)k0 + c_row) * DHEAD + dp;
      const float* vgp = V + bK + ((size_t)k0 + c_row) * DHEAD + dp;
      _Float16* kls = &Kl[c_row * KPITCH + dp];
      _Float16* vts = &VTl[dp * VPITCH + c_row];
#pragma unroll
      for (int i = 0; i < 16; ++i) {
        float2 kv = *(const float2*)(kgp + i * 256);      // +i*1024B const offset
        *(half2t*)(kls + i * 4 * KPITCH) = pack_f16(kv.x, kv.y);
        float2 vv = *(const float2*)(vgp + i * 256);
        vts[i * 4]          = (_Float16)vv.x;
        vts[VPITCH + i * 4] = (_Float16)vv.y;
      }
    }
    __syncthreads();

    // prefetch next tile (one 128B line per thread covers the 16KB f32 tile)
    {
      const int knext = k0 + TK;
      if (knext < Nk) {
        const char* kg2 = (const char*)(K + bK + (size_t)knext * DHEAD);
        const char* vg2 = (const char*)(V + bK + (size_t)knext * DHEAD);
        __builtin_prefetch(kg2 + tid * 128, 0, 3);
        __builtin_prefetch(vg2 + tid * 128, 0, 3);
      }
    }

    // ---- S^T = K * Q^T : 4 key-subtiles x 2 d-chunks = 8 WMMA ----
    // D layout: s[t][r] = S^T[key = k0 + t*16 + r + 8h][q = n]
    float s[4][8];
#pragma unroll
    for (int t = 0; t < 4; ++t) {
      half16 a0 = load_afrag128(kAbase + t * 16 * KPITCH + 0);
      half16 a1 = load_afrag128(kAbase + t * 16 * KPITCH + 32);
      f32x8 c = (f32x8){0.f,0.f,0.f,0.f,0.f,0.f,0.f,0.f};
      c = __builtin_amdgcn_wmma_f32_16x16x32_f16(false, a0, false, qf[0], (short)0, c, false, false);
      c = __builtin_amdgcn_wmma_f32_16x16x32_f16(false, a1, false, qf[1], (short)0, c, false, false);
#pragma unroll
      for (int r = 0; r < 8; ++r) {
        if (MASKED) {
          const int key = k0 + t * 16 + r + 8 * h;
          s[t][r] = (key < valid) ? c[r] : -1e6f;  // d2l masked_softmax NEG
        } else {
          s[t][r] = c[r];
        }
      }
    }

    // ---- online softmax: q row lives on lanes (n, n+16) -> one xor-16 combine ----
    float tmax = -INFINITY;
#pragma unroll
    for (int t = 0; t < 4; ++t)
#pragma unroll
      for (int r = 0; r < 8; ++r) tmax = fmaxf(tmax, s[t][r]);
    tmax = fmaxf(tmax, __shfl_xor(tmax, 16, 32));
    const float mnew  = fmaxf(mrow, tmax);
    const float alpha = __expf(mrow - mnew);   // first iter: exp(-inf)=0
    float tsum = 0.f;
    int pk[4][4];   // pk[t][j] = f16x2 of P for keys t*16 + 2j(+1) + 8h, q = n
#pragma unroll
    for (int t = 0; t < 4; ++t)
#pragma unroll
      for (int j = 0; j < 4; ++j) {
        const float e0 = __expf(s[t][2 * j]     - mnew);
        const float e1 = __expf(s[t][2 * j + 1] - mnew);
        tsum += e0 + e1;
        pk[t][j] = __builtin_bit_cast(int, pack_f16(e0, e1));
      }
    tsum += __shfl_xor(tsum, 16, 32);
    lrow = lrow * alpha + tsum;
    mrow = mnew;

    // acc rows are q = r + 8h -> broadcast alpha from lane (r + 8h)
#pragma unroll
    for (int r = 0; r < 8; ++r) {
      const float ar = __shfl(alpha, r + 8 * h, 32);
#pragma unroll
      for (int dt = 0; dt < 4; ++dt) acc[dt][r] *= ar;
    }

    // ---- O += P * V : A = P directly from registers (already in A layout) ----
    // A chunk kc pair j: j<4 -> pk[2kc][j], j>=4 -> pk[2kc+1][j-4]
#pragma unroll
    for (int kc = 0; kc < 2; ++kc) {
      half16 aP;
#pragma unroll
      for (int j = 0; j < 8; ++j) {
        half2t hv = __builtin_bit_cast(half2t, pk[2 * kc + (j >> 2)][j & 3]);
        aP[2 * j]     = hv[0];
        aP[2 * j + 1] = hv[1];
      }
#pragma unroll
      for (int dt = 0; dt < 4; ++dt) {
        half16 bV = load_bfrag(vBbase + dt * 16 * VPITCH + kc * 32);
        acc[dt] = __builtin_amdgcn_wmma_f32_16x16x32_f16(false, aP, false, bV, (short)0, acc[dt], false, false);
      }
    }
    __syncthreads();
  };

  for (int kt = 0; kt < nfull; ++kt) tile_body(kt * TK, MaskOff{});
  if (has_tail) tile_body(nfull * TK, MaskOn{});

  // Normalize and store: lane writes d = dt*16 + n at q = q0 + wave*16 + r + 8h (coalesced)
  const float invl = 1.0f / lrow;
  float* obase = O + bQ + ((size_t)q0 + wave * 16) * DHEAD;
#pragma unroll
  for (int r = 0; r < 8; ++r) {
    const float ir = __shfl(invl, r + 8 * h, 32);
    float* orow = obase + (size_t)(r + 8 * h) * DHEAD;
#pragma unroll
    for (int dt = 0; dt < 4; ++dt)
      orow[dt * 16 + n] = acc[dt][r] * ir;
  }
}

extern "C" void kernel_launch(void* const* d_in, const int* in_sizes, int n_in,
                              void* d_out, int out_size, void* d_ws, size_t ws_size,
                              hipStream_t stream) {
  const float* Q  = (const float*)d_in[0];
  const float* K  = (const float*)d_in[1];
  const float* V  = (const float*)d_in[2];
  const int*   vl = (const int*)d_in[3];
  float* O = (float*)d_out;
  const int B  = in_sizes[3];
  const int Nq = in_sizes[0] / (B * DHEAD);
  const int Nk = in_sizes[1] / (B * DHEAD);
  dim3 grid(Nq / TQ, B);
  flash_attn_wmma_f16<<<grid, 128, 0, stream>>>(Q, K, V, vl, O, Nq, Nk);
}
